// GoSpecificWattiPooling_62869731279402
// MI455X (gfx1250) — matrix operand order repaired and checked by reading, not compiled, not run on hardware
//
#include <hip/hip_runtime.h>

#define B_  16
#define L_  2048
#define T_  512
#define DH_ 1024
#define DG_ 768
#define P_  256

#define CH     32     // l-chunk (WMMA K depth for Z-GEMM)
#define SQ_STR 264    // LDS row stride (shorts) for Q tile
#define SP_STR 40     // LDS row stride (shorts) for P tile
#define SK_STR 264    // LDS row stride (shorts) for K tile (pad: conflict-free b128 reads)

typedef __attribute__((ext_vector_type(16))) __bf16       v16bf;
typedef __attribute__((ext_vector_type(2)))  __bf16       v2bf;
typedef __attribute__((ext_vector_type(8)))  float        v8f;
typedef __attribute__((ext_vector_type(2)))  float        v2f;
typedef __attribute__((ext_vector_type(8)))  unsigned int v8u;

__device__ __forceinline__ v8f zero8() {
  v8f z = {0.f,0.f,0.f,0.f,0.f,0.f,0.f,0.f};
  return z;
}

__device__ __forceinline__ v8f wmma_bf16(v8u a, v8u b, v8f c) {
  return __builtin_amdgcn_wmma_f32_16x16x32_bf16(
      false, __builtin_bit_cast(v16bf, a),
      false, __builtin_bit_cast(v16bf, b),
      (short)0, c, false, false);
}

// native packed f32 -> bf16 (RNE): v_cvt_pk_bf16_f32
__device__ __forceinline__ unsigned int pack2bf(float a, float b) {
  v2f f = {a, b};
  return __builtin_bit_cast(unsigned int, __builtin_convertvector(f, v2bf));
}
__device__ __forceinline__ unsigned short f2bf(float a) {
  return __builtin_bit_cast(unsigned short, (__bf16)a);
}

// A fragment (16x32 bf16): lane holds row (lane&15); k = e + 16*h (contiguous)
__device__ __forceinline__ v8u ld_a16(const unsigned short* row, int h) {
  uint4 a = *(const uint4*)(row + 16*h);
  uint4 b = *(const uint4*)(row + 16*h + 8);
  v8u r = {a.x,a.y,a.z,a.w,b.x,b.y,b.z,b.w};
  return r;
}
// B fragment (32x16 bf16): lane holds column (lane&15); k = e + 16*h (contiguous)
__device__ __forceinline__ v8u ld_b16(const unsigned short* col, int h) {
  const uint4* q = (const uint4*)(col + 16*h);
  uint4 a = q[0], b = q[1];
  v8u r = {a.x,a.y,a.z,a.w,b.x,b.y,b.z,b.w};
  return r;
}
// fragments loaded from fp32 global with packed conversion
__device__ __forceinline__ v8u cvt_a32(const float* row, int h) {
  const float4* q0 = (const float4*)(row + 16*h);
  float4 a0 = q0[0], a1 = q0[1], b0 = q0[2], b1 = q0[3];
  v8u r = { pack2bf(a0.x,a0.y), pack2bf(a0.z,a0.w), pack2bf(a1.x,a1.y), pack2bf(a1.z,a1.w),
            pack2bf(b0.x,b0.y), pack2bf(b0.z,b0.w), pack2bf(b1.x,b1.y), pack2bf(b1.z,b1.w) };
  return r;
}
__device__ __forceinline__ v8u cvt_b32(const float* col, int h) {
  const float4* q = (const float4*)(col + 16*h);
  float4 a = q[0], b = q[1], c = q[2], d = q[3];
  v8u r = { pack2bf(a.x,a.y), pack2bf(a.z,a.w), pack2bf(b.x,b.y), pack2bf(b.z,b.w),
            pack2bf(c.x,c.y), pack2bf(c.z,c.w), pack2bf(d.x,d.y), pack2bf(d.z,d.w) };
  return r;
}

// Async DMA: each lane copies its own 512B K-row in 32 x 16B chunks.
// offset: immediate is added to BOTH the global and LDS addresses (ISA §10),
// so base registers are loop-invariant (single VGPR pair + one LDS VGPR).
__device__ __forceinline__ void async_copy_k(const unsigned short* gsrc_lane,
                                             unsigned lds_lane) {
  #pragma unroll
  for (int i = 0; i < 32; ++i) {
    asm volatile("global_load_async_to_lds_b128 %0, %1, off offset:%2"
                 :: "v"(lds_lane), "v"(gsrc_lane), "i"(i * 16) : "memory");
  }
}
__device__ __forceinline__ void wait_async0() {
  asm volatile("s_wait_asynccnt 0x0" ::: "memory");
}

// ---------------------------------------------------------------------------
// One-time: H (B,L,DH) fp32 -> HbT (B,DH,L) bf16, 64x64 LDS tile transpose.
// ---------------------------------------------------------------------------
__global__ __launch_bounds__(256) void gswp_cvth(const float* __restrict__ H,
                                                 unsigned short* __restrict__ HbT)
{
  __shared__ unsigned int sT[64 * 33];   // [l][d/2] packed bf16 pairs, pad 1 dword
  const int b   = blockIdx.z;
  const int l0  = blockIdx.x * 64;
  const int d0  = blockIdx.y * 64;
  const int tid = threadIdx.x;

  const float* src = H + ((size_t)b * L_ + l0) * DH_ + d0;
  #pragma unroll
  for (int it = 0; it < 4; ++it) {
    int idx = tid + it * 256;            // [0,1024) float4 slots
    int l = idx >> 4, c4 = idx & 15;
    float4 v = ((const float4*)(src + (size_t)l * DH_))[c4];
    sT[l * 33 + c4 * 2 + 0] = pack2bf(v.x, v.y);
    sT[l * 33 + c4 * 2 + 1] = pack2bf(v.z, v.w);
  }
  __syncthreads();

  unsigned short* dst = HbT + ((size_t)b * DH_ + d0) * L_ + l0;
  #pragma unroll
  for (int it = 0; it < 2; ++it) {
    int idx = tid + it * 256;            // [0,512) uint4 slots
    int d = idx >> 3, lg = (idx & 7) * 8;
    unsigned e[8];
    #pragma unroll
    for (int j = 0; j < 8; ++j) {
      unsigned w = sT[(lg + j) * 33 + (d >> 1)];
      e[j] = (d & 1) ? (w >> 16) : (w & 0xFFFFu);
    }
    uint4 o = { e[0] | (e[1] << 16), e[2] | (e[3] << 16),
                e[4] | (e[5] << 16), e[6] | (e[7] << 16) };
    *((uint4*)(dst + (size_t)d * L_ + lg)) = o;
  }
}

// ---------------------------------------------------------------------------
// Out[m][n] = sum_k X[m][k]*W[n][k], bf16 out (M x 256).
// One wave per 16x64 block: A fragment reused across 4 n-tiles.
// ---------------------------------------------------------------------------
__global__ __launch_bounds__(128) void gswp_proj(
    const float* __restrict__ X, const float* __restrict__ W,
    unsigned short* __restrict__ Out, int M, int Kd)
{
  const int lane = threadIdx.x & 31;
  const int wave = threadIdx.x >> 5;
  const int wid  = blockIdx.x * 4 + wave;
  const int m0   = (wid >> 2) * 16;
  const int n0   = (wid & 3) * 64;
  const int h    = lane >> 4;
  const int ln   = lane & 15;

  const float* xrow = X + (size_t)(m0 + ln) * Kd;
  const float* w0   = W + (size_t)(n0 + ln) * Kd;

  v8f acc0 = zero8(), acc1 = zero8(), acc2 = zero8(), acc3 = zero8();
  for (int k0 = 0; k0 < Kd; k0 += 32) {
    v8u a = cvt_a32(xrow + k0, h);
    acc0 = wmma_bf16(a, cvt_b32(w0 + k0, h), acc0);
    acc1 = wmma_bf16(a, cvt_b32(w0 + (size_t)16 * Kd + k0, h), acc1);
    acc2 = wmma_bf16(a, cvt_b32(w0 + (size_t)32 * Kd + k0, h), acc2);
    acc3 = wmma_bf16(a, cvt_b32(w0 + (size_t)48 * Kd + k0, h), acc3);
  }
  unsigned short* orow = Out + (size_t)(m0 + 8*h) * P_ + n0 + ln;
  #pragma unroll
  for (int r = 0; r < 8; ++r) {
    orow[(size_t)r * P_ +  0] = f2bf(acc0[r]);
    orow[(size_t)r * P_ + 16] = f2bf(acc1[r]);
    orow[(size_t)r * P_ + 32] = f2bf(acc2[r]);
    orow[(size_t)r * P_ + 48] = f2bf(acc3[r]);
  }
}

// ---------------------------------------------------------------------------
// Flash attention: one block = (batch b, 16 query rows), 8 waves.
// Wave 0: async-double-buffered K tile in LDS -> S = Q.K^T + online softmax
// -> P (LDS). All waves: Z += P.H via B-fragments loaded straight from
// transposed bf16 H in global (L2-resident, prefetched a chunk ahead).
// ---------------------------------------------------------------------------
__global__ __launch_bounds__(256) void gswp_attn(
    const unsigned short* __restrict__ HbT, const unsigned short* __restrict__ Kbf,
    const unsigned short* __restrict__ Qbf, const unsigned char* __restrict__ mask,
    float* __restrict__ out)
{
  __shared__ unsigned short sK[2 * 32 * SK_STR];   // double-buffered K tile
  __shared__ unsigned short sQ[16 * SQ_STR];
  __shared__ unsigned short sP[16 * SP_STR];
  __shared__ float sCorr[16];
  __shared__ float sSum[16];

  const int b    = blockIdx.y;
  const int t0   = blockIdx.x * 16;
  const int tid  = threadIdx.x;
  const int lane = tid & 31;
  const int wave = tid >> 5;
  const int h    = lane >> 4;
  const int ln   = lane & 15;
  const float NINF = -__builtin_inff();

  { // Q tile (16 x 256 bf16) -> LDS once
    const unsigned int* qs = (const unsigned int*)(Qbf + (size_t)(b * T_ + t0) * P_);
    for (int i = tid; i < 16 * (P_/2); i += 256) {
      int r = i >> 7, c = i & 127;
      ((unsigned int*)sQ)[r * (SQ_STR/2) + c] = qs[r * (P_/2) + c];
    }
  }
  if (wave == 0) {  // kick off async copy of the first K tile (buffer 0)
    async_copy_k(Kbf + (size_t)(b * L_ + lane) * P_,
                 (unsigned)(size_t)(void*)(sK + lane * SK_STR));
  }
  __syncthreads();

  v8f acc[8];
  #pragma unroll
  for (int i = 0; i < 8; ++i) acc[i] = zero8();
  float mrun[8], rsum[8];
  #pragma unroll
  for (int r = 0; r < 8; ++r) { mrun[r] = NINF; rsum[r] = 0.f; }

  const float scale = 0.0625f;               // 1/sqrt(256)
  const float LOG2E = 1.4426950408889634f;
  const unsigned short* hbt = HbT + (size_t)b * DH_ * L_;

  for (int ci = 0; ci < L_ / CH; ++ci) {
    const int l0 = ci * CH;

    if (wave == 0) {
      // ---- wait for this chunk's K tile; immediately launch the next one
      wait_async0();
      if (ci + 1 < L_ / CH) {
        async_copy_k(Kbf + (size_t)(b * L_ + l0 + CH + lane) * P_,
                     (unsigned)(size_t)(void*)
                         (sK + ((ci + 1) & 1) * (32 * SK_STR) + lane * SK_STR));
      }
      const unsigned short* kb = sK + (ci & 1) * (32 * SK_STR);

      // ---- S = Q . K^T (two 16x16 frags), then online softmax
      v8f s0 = zero8(), s1 = zero8();
      #pragma unroll
      for (int pc = 0; pc < 8; ++pc) {
        v8u aq = ld_a16(sQ + ln * SQ_STR + pc * 32, h);
        v8u b0 = ld_b16(kb + ln * SK_STR + pc * 32, h);
        v8u b1 = ld_b16(kb + (16 + ln) * SK_STR + pc * 32, h);
        s0 = wmma_bf16(aq, b0, s0);
        s1 = wmma_bf16(aq, b1, s1);
      }
      const bool mk0 = mask[(size_t)b * L_ + l0 + ln] != 0;
      const bool mk1 = mask[(size_t)b * L_ + l0 + 16 + ln] != 0;
      #pragma unroll
      for (int r = 0; r < 8; ++r) {
        float v0 = mk0 ? NINF : s0[r] * scale;
        float v1 = mk1 ? NINF : s1[r] * scale;
        float mx = fmaxf(v0, v1);
        #pragma unroll
        for (int o = 1; o < 16; o <<= 1) mx = fmaxf(mx, __shfl_xor(mx, o, 32));
        float mnew = fmaxf(mrun[r], mx);
        float corr = (mnew == NINF) ? 0.f : exp2f((mrun[r] - mnew) * LOG2E);
        float p0 = mk0 ? 0.f : exp2f((v0 - mnew) * LOG2E);
        float p1 = mk1 ? 0.f : exp2f((v1 - mnew) * LOG2E);
        float ps = p0 + p1;
        #pragma unroll
        for (int o = 1; o < 16; o <<= 1) ps += __shfl_xor(ps, o, 32);
        rsum[r] = rsum[r] * corr + ps;
        mrun[r] = mnew;
        sP[(r + 8*h) * SP_STR + ln]      = f2bf(p0);
        sP[(r + 8*h) * SP_STR + 16 + ln] = f2bf(p1);
        if (ln == 0) sCorr[r + 8*h] = corr;
      }
    } else {
      // ---- idle waves prefetch next chunk's H columns (one line per d-row)
      #pragma unroll
      for (int nt = 0; nt < 8; ++nt) {
        const unsigned short* p =
            hbt + (size_t)(wave * 128 + nt * 16 + ln) * L_ + l0 + CH;
        __builtin_prefetch((const void*)p, 0, 1);
      }
    }
    __syncthreads();

    // ---- all waves: rescale accumulators, then Z += P . H
    float cv[8];
    #pragma unroll
    for (int r = 0; r < 8; ++r) cv[r] = sCorr[r + 8*h];
    #pragma unroll
    for (int nt = 0; nt < 8; ++nt)
      #pragma unroll
      for (int r = 0; r < 8; ++r) acc[nt][r] *= cv[r];

    v8u ap = ld_a16(sP + ln * SP_STR, h);
    #pragma unroll
    for (int nt = 0; nt < 8; ++nt) {
      const unsigned short* hcol =
          hbt + (size_t)(wave * 128 + nt * 16 + ln) * L_ + l0;
      acc[nt] = wmma_bf16(ap, ld_b16(hcol, h), acc[nt]);
    }
    __syncthreads();
  }

  // ---- finalize: divide by softmax denominator, store fp32
  if (wave == 0 && ln == 0) {
    #pragma unroll
    for (int r = 0; r < 8; ++r) sSum[r + 8*h] = rsum[r];
  }
  __syncthreads();
  #pragma unroll
  for (int r = 0; r < 8; ++r) {
    float inv = 1.0f / sSum[r + 8*h];
    float* orow = out + (size_t)(b * T_ + t0 + r + 8*h) * DH_;
    #pragma unroll
    for (int nt = 0; nt < 8; ++nt)
      orow[wave * 128 + nt * 16 + ln] = acc[nt][r] * inv;
  }
}

extern "C" void kernel_launch(void* const* d_in, const int* in_sizes, int n_in,
                              void* d_out, int out_size, void* d_ws, size_t ws_size,
                              hipStream_t stream) {
  const float* H  = (const float*)d_in[0];
  const float* G  = (const float*)d_in[1];
  const unsigned char* mask = (const unsigned char*)d_in[2];
  const float* Wk = (const float*)d_in[3];
  const float* Wq = (const float*)d_in[4];
  float* out = (float*)d_out;

  unsigned short* Kbf = (unsigned short*)d_ws;                  // B*L*256  bf16 (16.8 MB)
  unsigned short* Qbf = Kbf + (size_t)B_ * L_ * P_;             // B*T*256  bf16 ( 4.2 MB)
  unsigned short* HbT = Qbf + (size_t)B_ * T_ * P_;             // B*DH*L   bf16 (67.1 MB)

  gswp_cvth<<<dim3(L_ / 64, DH_ / 64, B_), 256, 0, stream>>>(H, HbT);
  gswp_proj<<<dim3((B_ * L_) / 16), 128, 0, stream>>>(H, Wk, Kbf, B_ * L_, DH_);
  gswp_proj<<<dim3((B_ * T_) / 16), 128, 0, stream>>>(G, Wq, Qbf, B_ * T_, DG_);
  gswp_attn<<<dim3(T_ / 16, B_), 256, 0, stream>>>(HbT, Kbf, Qbf, mask, out);
}
